// NodeGNN_hyunmok_bp_12395275616821
// MI455X (gfx1250) — matrix operand order repaired and checked by reading, compiled, not run on hardware
//
#include <hip/hip_runtime.h>
#include <hip/hip_bf16.h>

#define NODES 50000
#define EDGES 800000
#define FP 160          // feature K padded 136 -> 160 (mult of 32)
#define OKP 96          // out-mlp K padded 66 -> 96

typedef __attribute__((ext_vector_type(16))) __bf16 v16bf;
typedef __attribute__((ext_vector_type(8)))  __bf16 v8bf;
typedef __attribute__((ext_vector_type(8)))  float  v8f;

__device__ __forceinline__ v8f wmma_bf16(v16bf a, v16bf b, v8f c) {
  return __builtin_amdgcn_wmma_f32_16x16x32_bf16(false, a, false, b, (short)0, c, false, false);
}

// A-fragment (16x32 bf16) per ISA layout: lane = M (0..15) + 16*khalf.
// elems 0..7  -> K = kbase + 0..7        (kbase = kt*32 + 8*khalf)
// elems 8..15 -> K = kbase + 16 + 0..7
__device__ __forceinline__ v16bf load_a_frag(const __bf16* row, int kbase) {
  v8bf a0 = *(const v8bf*)(row + kbase);
  v8bf a1 = *(const v8bf*)(row + kbase + 16);
  v16bf a;
#pragma unroll
  for (int i = 0; i < 8; i++) { a[i] = a0[i]; a[i + 8] = a1[i]; }
  return a;
}

// B fragments pre-packed: frag blob = 32 lanes x 16 bf16 contiguous
__device__ __forceinline__ v16bf load_b_frag(const __bf16* frags, int frag, int lane) {
  return *(const v16bf*)(frags + (size_t)frag * 512 + lane * 16);
}

__device__ __forceinline__ float sigf(float x) { return 1.f / (1.f + __expf(-x)); }

// ---------------------------------------------------------------------------
// Pack fp32 weight (Kreal x Nreal, row-major) into WMMA B fragments, zero-padded
// to KP x NP.  B layout: lane = khalf * 16 + N%16 ; elem e -> K offset 8*khalf + (e<8 ? e : 8+e).
// ---------------------------------------------------------------------------
__global__ void pack_b_frags(const float* __restrict__ src, __bf16* __restrict__ dst,
                             int Kreal, int Nreal, int KP, int NP) {
  int ntiles = NP / 16;
  int frag = blockIdx.x;
  int kt = frag / ntiles, nt = frag % ntiles;
  int lane = threadIdx.x >> 4;   // 0..31
  int e    = threadIdx.x & 15;   // 0..15
  int K = kt * 32 + (lane >> 4) * 8 + (e < 8 ? e : 8 + e);
  int N = nt * 16 + (lane & 15);
  float v = (K < Kreal && N < Nreal) ? src[(size_t)K * Nreal + N] : 0.f;
  dst[(size_t)frag * 512 + lane * 16 + e] = (__bf16)v;
}

// ---------------------------------------------------------------------------
// Edge kernel: per wave = 16 edges. Gather feat -> LDS bf16, msg MLP (3 layers)
// and att head via WMMA, write msg (bf16), att, atomic att_sum.
// ---------------------------------------------------------------------------
__global__ void __launch_bounds__(256)
edge_kernel(const float* __restrict__ state, const float* __restrict__ bnode,
            const float* __restrict__ Jmsg, const int* __restrict__ mnode,
            const __bf16* W1f, const float* b1, const __bf16* W2f, const float* b2,
            const __bf16* W3f, const float* b3,
            const __bf16* aW1f, const float* ab1,
            const __bf16* aW2f, const float* ab2,
            __bf16* __restrict__ msg_buf, float* __restrict__ att_buf,
            float* __restrict__ att_sum) {
  __shared__ __bf16 sfeat[8][16 * FP];
  __shared__ __bf16 sh[8][16 * 64];
  const int wave = threadIdx.x >> 5, lane = threadIdx.x & 31;
  const int n = lane & 15, kh = lane >> 4;
  const int tile = blockIdx.x * 8 + wave;
  const int e0 = tile * 16;

  { // cooperative gather: lane pair (n, n+16) fills row n halves
    const int row = n, half = kh;
    const int e = e0 + row;
    const bool ok = e < EDGES;
    const float J = ok ? Jmsg[e] : 0.f;
    __bf16* dr = &sfeat[wave][row * FP];
    if (half == 0) {
      const int vi = ok ? mnode[2 * e] : 0;
      const float4* s4 = (const float4*)(state + (size_t)vi * 64);
#pragma unroll
      for (int i = 0; i < 16; i++) {
        float4 vv = ok ? s4[i] : make_float4(0.f, 0.f, 0.f, 0.f);
        dr[4 * i + 0] = (__bf16)vv.x; dr[4 * i + 1] = (__bf16)vv.y;
        dr[4 * i + 2] = (__bf16)vv.z; dr[4 * i + 3] = (__bf16)vv.w;
      }
      const float bi = ok ? bnode[vi] : 0.f;
      dr[64] = (__bf16)bi; dr[65] = (__bf16)(-bi);
      dr[66] = (__bf16)J;  dr[67] = (__bf16)(-J);
    } else {
      const int vo = ok ? mnode[2 * e + 1] : 0;
      const float4* s4 = (const float4*)(state + (size_t)vo * 64);
#pragma unroll
      for (int i = 0; i < 16; i++) {
        float4 vv = ok ? s4[i] : make_float4(0.f, 0.f, 0.f, 0.f);
        dr[68 + 4 * i + 0] = (__bf16)vv.x; dr[68 + 4 * i + 1] = (__bf16)vv.y;
        dr[68 + 4 * i + 2] = (__bf16)vv.z; dr[68 + 4 * i + 3] = (__bf16)vv.w;
      }
      const float bo = ok ? bnode[vo] : 0.f;
      dr[132] = (__bf16)(-bo); dr[133] = (__bf16)bo;
      dr[134] = (__bf16)(-J);  dr[135] = (__bf16)J;
#pragma unroll
      for (int c2 = 136; c2 < FP; c2++) dr[c2] = (__bf16)0.f;
    }
  }
  __syncthreads();

  v16bf afeat[5];
#pragma unroll
  for (int kt = 0; kt < 5; kt++) afeat[kt] = load_a_frag(&sfeat[wave][n * FP], kt * 32 + 8 * kh);

  // msg layer 1 (160 -> 64), relu -> sh
#pragma unroll
  for (int nt = 0; nt < 4; nt++) {
    v8f c; const float bb = b1[nt * 16 + n];
#pragma unroll
    for (int r = 0; r < 8; r++) c[r] = bb;
#pragma unroll
    for (int kt = 0; kt < 5; kt++) c = wmma_bf16(afeat[kt], load_b_frag(W1f, kt * 4 + nt, lane), c);
#pragma unroll
    for (int r = 0; r < 8; r++) sh[wave][(r + 8 * kh) * 64 + nt * 16 + n] = (__bf16)fmaxf(c[r], 0.f);
  }
  __syncthreads();
  v16bf ah0 = load_a_frag(&sh[wave][n * 64], 8 * kh);
  v16bf ah1 = load_a_frag(&sh[wave][n * 64], 32 + 8 * kh);
  __syncthreads();

  // msg layer 2 (64 -> 64), relu -> sh
#pragma unroll
  for (int nt = 0; nt < 4; nt++) {
    v8f c; const float bb = b2[nt * 16 + n];
#pragma unroll
    for (int r = 0; r < 8; r++) c[r] = bb;
    c = wmma_bf16(ah0, load_b_frag(W2f, nt, lane), c);
    c = wmma_bf16(ah1, load_b_frag(W2f, 4 + nt, lane), c);
#pragma unroll
    for (int r = 0; r < 8; r++) sh[wave][(r + 8 * kh) * 64 + nt * 16 + n] = (__bf16)fmaxf(c[r], 0.f);
  }
  __syncthreads();
  ah0 = load_a_frag(&sh[wave][n * 64], 8 * kh);
  ah1 = load_a_frag(&sh[wave][n * 64], 32 + 8 * kh);

  // msg layer 3 (64 -> 64), store bf16 to global
#pragma unroll
  for (int nt = 0; nt < 4; nt++) {
    v8f c; const float bb = b3[nt * 16 + n];
#pragma unroll
    for (int r = 0; r < 8; r++) c[r] = bb;
    c = wmma_bf16(ah0, load_b_frag(W3f, nt, lane), c);
    c = wmma_bf16(ah1, load_b_frag(W3f, 4 + nt, lane), c);
#pragma unroll
    for (int r = 0; r < 8; r++) {
      int e = e0 + r + 8 * kh;
      if (e < EDGES) msg_buf[(size_t)e * 64 + nt * 16 + n] = (__bf16)c[r];
    }
  }
  __syncthreads();

  // att layer 1 (reuse cached feat A-frags), relu -> sh
#pragma unroll
  for (int nt = 0; nt < 4; nt++) {
    v8f c; const float bb = ab1[nt * 16 + n];
#pragma unroll
    for (int r = 0; r < 8; r++) c[r] = bb;
#pragma unroll
    for (int kt = 0; kt < 5; kt++) c = wmma_bf16(afeat[kt], load_b_frag(aW1f, kt * 4 + nt, lane), c);
#pragma unroll
    for (int r = 0; r < 8; r++) sh[wave][(r + 8 * kh) * 64 + nt * 16 + n] = (__bf16)fmaxf(c[r], 0.f);
  }
  __syncthreads();
  ah0 = load_a_frag(&sh[wave][n * 64], 8 * kh);
  ah1 = load_a_frag(&sh[wave][n * 64], 32 + 8 * kh);

  // att layer 2: 64 -> 1 (N padded to 16, column 0 valid)
  {
    v8f c; const float bb = ab2[0];
#pragma unroll
    for (int r = 0; r < 8; r++) c[r] = bb;
    c = wmma_bf16(ah0, load_b_frag(aW2f, 0, lane), c);
    c = wmma_bf16(ah1, load_b_frag(aW2f, 1, lane), c);
    if (n == 0) {
#pragma unroll
      for (int r = 0; r < 8; r++) {
        int e = e0 + r + 8 * kh;
        if (e < EDGES) {
          float att = __expf(sigf(c[r]));
          att_buf[e] = att;
          int vo = mnode[2 * e + 1];
          unsafeAtomicAdd(&att_sum[vo], att);
        }
      }
    }
  }
}

// ---------------------------------------------------------------------------
// Scatter: normalized attention-weighted aggregation. 1 thread per (edge, ch).
// ---------------------------------------------------------------------------
__global__ void __launch_bounds__(256)
scatter_kernel(const __bf16* __restrict__ msg_buf, const float* __restrict__ att_buf,
               const float* __restrict__ att_sum, const int* __restrict__ mnode,
               float* __restrict__ msg_agg) {
  size_t tid = (size_t)blockIdx.x * 256 + threadIdx.x;
  if (tid >= (size_t)EDGES * 64) return;
  int e = (int)(tid >> 6);
  int c = (int)(tid & 63);
  int vo = mnode[2 * e + 1];
  float w = att_buf[e] / att_sum[vo];
  unsafeAtomicAdd(&msg_agg[(size_t)vo * 64 + c], w * (float)msg_buf[tid]);
}

// ---------------------------------------------------------------------------
// Node kernel: GRU + output MLP, per wave = 16 nodes.
// ---------------------------------------------------------------------------
__global__ void __launch_bounds__(256)
node_kernel(const float* __restrict__ state, float* __restrict__ state_new,
            const float* __restrict__ msg_agg, const float* __restrict__ bnode,
            const __bf16* Wihf, const float* bih, const __bf16* Whhf, const float* bhh,
            const __bf16* oW1f, const float* ob1, const __bf16* oW2f, const float* ob2,
            const __bf16* oW3f, const float* ob3,
            float* __restrict__ out, int step) {
  __shared__ __bf16 sbuf[8][16 * OKP];  // 16x64 (msg/state tiles) then 16x96 (out-mlp input)
  __shared__ __bf16 sh[8][16 * 64];
  __shared__ float  sy[8][16 * 2];
  const int wave = threadIdx.x >> 5, lane = threadIdx.x & 31;
  const int n = lane & 15, kh = lane >> 4;
  const int tile = blockIdx.x * 8 + wave;
  const int v0 = tile * 16;

  // ---- msg_agg tile -> sbuf (stride OKP), A-frags
  {
    int vtx = v0 + n; bool ok = vtx < NODES;
    const float4* s4 = (const float4*)(msg_agg + (size_t)vtx * 64 + kh * 32);
    __bf16* dr = &sbuf[wave][n * OKP + kh * 32];
#pragma unroll
    for (int i = 0; i < 8; i++) {
      float4 vv = ok ? s4[i] : make_float4(0.f, 0.f, 0.f, 0.f);
      dr[4 * i + 0] = (__bf16)vv.x; dr[4 * i + 1] = (__bf16)vv.y;
      dr[4 * i + 2] = (__bf16)vv.z; dr[4 * i + 3] = (__bf16)vv.w;
    }
  }
  __syncthreads();
  v16bf am0 = load_a_frag(&sbuf[wave][n * OKP], 8 * kh);
  v16bf am1 = load_a_frag(&sbuf[wave][n * OKP], 32 + 8 * kh);
  __syncthreads();
  // ---- state tile -> sbuf
  {
    int vtx = v0 + n; bool ok = vtx < NODES;
    const float4* s4 = (const float4*)(state + (size_t)vtx * 64 + kh * 32);
    __bf16* dr = &sbuf[wave][n * OKP + kh * 32];
#pragma unroll
    for (int i = 0; i < 8; i++) {
      float4 vv = ok ? s4[i] : make_float4(0.f, 0.f, 0.f, 0.f);
      dr[4 * i + 0] = (__bf16)vv.x; dr[4 * i + 1] = (__bf16)vv.y;
      dr[4 * i + 2] = (__bf16)vv.z; dr[4 * i + 3] = (__bf16)vv.w;
    }
  }
  __syncthreads();
  v16bf as0 = load_a_frag(&sbuf[wave][n * OKP], 8 * kh);
  v16bf as1 = load_a_frag(&sbuf[wave][n * OKP], 32 + 8 * kh);

  // ---- GRU matmuls: r,z gates fused (gi+gh), n-parts separate
  v8f rz[8], inn[4], hnn[4];
#pragma unroll
  for (int nt = 0; nt < 8; nt++) {
    v8f c; const float bb = bih[nt * 16 + n] + bhh[nt * 16 + n];
#pragma unroll
    for (int r = 0; r < 8; r++) c[r] = bb;
    c = wmma_bf16(am0, load_b_frag(Wihf, nt, lane), c);
    c = wmma_bf16(am1, load_b_frag(Wihf, 12 + nt, lane), c);
    c = wmma_bf16(as0, load_b_frag(Whhf, nt, lane), c);
    c = wmma_bf16(as1, load_b_frag(Whhf, 12 + nt, lane), c);
    rz[nt] = c;
  }
#pragma unroll
  for (int nt = 0; nt < 4; nt++) {
    v8f c; const float bb = bih[128 + nt * 16 + n];
#pragma unroll
    for (int r = 0; r < 8; r++) c[r] = bb;
    c = wmma_bf16(am0, load_b_frag(Wihf, 8 + nt, lane), c);
    c = wmma_bf16(am1, load_b_frag(Wihf, 20 + nt, lane), c);
    inn[nt] = c;
  }
#pragma unroll
  for (int nt = 0; nt < 4; nt++) {
    v8f c; const float bb = bhh[128 + nt * 16 + n];
#pragma unroll
    for (int r = 0; r < 8; r++) c[r] = bb;
    c = wmma_bf16(as0, load_b_frag(Whhf, 8 + nt, lane), c);
    c = wmma_bf16(as1, load_b_frag(Whhf, 20 + nt, lane), c);
    hnn[nt] = c;
  }

  // ---- gates + state update (in-place: each (M,h) owned by one lane)
#pragma unroll
  for (int nt = 0; nt < 4; nt++) {
#pragma unroll
    for (int r = 0; r < 8; r++) {
      int M = r + 8 * kh, h = nt * 16 + n, vtx = v0 + M;
      float rg = sigf(rz[nt][r]);
      float zg = sigf(rz[nt + 4][r]);
      float ng = tanhf(inn[nt][r] + rg * hnn[nt][r]);
      float s = (float)sbuf[wave][M * OKP + h];
      float sn = (1.f - zg) * ng + zg * s;
      if (vtx < NODES) state_new[(size_t)vtx * 64 + h] = sn;
      sbuf[wave][M * OKP + h] = (__bf16)sn;
    }
  }
  if (n == 0) {  // out-mlp extra inputs: [b, -b, 0-pad] -> cols 64..95
#pragma unroll
    for (int r = 0; r < 8; r++) {
      int M = r + 8 * kh, vtx = v0 + M;
      float bv = vtx < NODES ? bnode[vtx] : 0.f;
      __bf16* dr = &sbuf[wave][M * OKP];
      dr[64] = (__bf16)bv; dr[65] = (__bf16)(-bv);
#pragma unroll
      for (int c2 = 66; c2 < OKP; c2++) dr[c2] = (__bf16)0.f;
    }
  }
  __syncthreads();

  v16bf ao[3];
#pragma unroll
  for (int kt = 0; kt < 3; kt++) ao[kt] = load_a_frag(&sbuf[wave][n * OKP], kt * 32 + 8 * kh);

  // out layer 1 (96 -> 64), relu -> sh
#pragma unroll
  for (int nt = 0; nt < 4; nt++) {
    v8f c; const float bb = ob1[nt * 16 + n];
#pragma unroll
    for (int r = 0; r < 8; r++) c[r] = bb;
#pragma unroll
    for (int kt = 0; kt < 3; kt++) c = wmma_bf16(ao[kt], load_b_frag(oW1f, kt * 4 + nt, lane), c);
#pragma unroll
    for (int r = 0; r < 8; r++) sh[wave][(r + 8 * kh) * 64 + nt * 16 + n] = (__bf16)fmaxf(c[r], 0.f);
  }
  __syncthreads();
  v16bf ah0 = load_a_frag(&sh[wave][n * 64], 8 * kh);
  v16bf ah1 = load_a_frag(&sh[wave][n * 64], 32 + 8 * kh);
  __syncthreads();
  // out layer 2 (64 -> 64), relu -> sh
#pragma unroll
  for (int nt = 0; nt < 4; nt++) {
    v8f c; const float bb = ob2[nt * 16 + n];
#pragma unroll
    for (int r = 0; r < 8; r++) c[r] = bb;
    c = wmma_bf16(ah0, load_b_frag(oW2f, nt, lane), c);
    c = wmma_bf16(ah1, load_b_frag(oW2f, 4 + nt, lane), c);
#pragma unroll
    for (int r = 0; r < 8; r++) sh[wave][(r + 8 * kh) * 64 + nt * 16 + n] = (__bf16)fmaxf(c[r], 0.f);
  }
  __syncthreads();
  ah0 = load_a_frag(&sh[wave][n * 64], 8 * kh);
  ah1 = load_a_frag(&sh[wave][n * 64], 32 + 8 * kh);

  // out layer 3 (64 -> 2, N padded 16): p0 = exp(log_softmax)[0] = sigmoid(y0 - y1)
  {
    v8f c; const float bb = (n < 2) ? ob3[n] : 0.f;
#pragma unroll
    for (int r = 0; r < 8; r++) c[r] = bb;
    c = wmma_bf16(ah0, load_b_frag(oW3f, 0, lane), c);
    c = wmma_bf16(ah1, load_b_frag(oW3f, 1, lane), c);
    if (n < 2) {
#pragma unroll
      for (int r = 0; r < 8; r++) sy[wave][(r + 8 * kh) * 2 + n] = c[r];
    }
  }
  __syncthreads();
  if (lane < 16) {
    int vtx = v0 + lane;
    if (vtx < NODES) {
      float y0 = sy[wave][lane * 2 + 0], y1 = sy[wave][lane * 2 + 1];
      out[(size_t)vtx * 3 + step] = sigf(y0 - y1);
    }
  }
}

// ---------------------------------------------------------------------------
extern "C" void kernel_launch(void* const* d_in, const int* in_sizes, int n_in,
                              void* d_out, int out_size, void* d_ws, size_t ws_size,
                              hipStream_t stream) {
  const float* J_msg  = (const float*)d_in[0];
  const float* b      = (const float*)d_in[1];
  const int*   mnode  = (const int*)d_in[2];
  const float* msg_W1 = (const float*)d_in[4];  const float* msg_b1 = (const float*)d_in[5];
  const float* msg_W2 = (const float*)d_in[6];  const float* msg_b2 = (const float*)d_in[7];
  const float* msg_W3 = (const float*)d_in[8];  const float* msg_b3 = (const float*)d_in[9];
  const float* att_W1 = (const float*)d_in[10]; const float* att_b1 = (const float*)d_in[11];
  const float* att_W2 = (const float*)d_in[12]; const float* att_b2 = (const float*)d_in[13];
  const float* gru_Wih = (const float*)d_in[14]; const float* gru_bih = (const float*)d_in[15];
  const float* gru_Whh = (const float*)d_in[16]; const float* gru_bhh = (const float*)d_in[17];
  const float* out_W1 = (const float*)d_in[18]; const float* out_b1 = (const float*)d_in[19];
  const float* out_W2 = (const float*)d_in[20]; const float* out_b2 = (const float*)d_in[21];
  const float* out_W3 = (const float*)d_in[22]; const float* out_b3 = (const float*)d_in[23];

  char* w = (char*)d_ws;
  size_t off = 0;
  auto carve = [&](size_t bytes) { void* p = w + off; off = (off + bytes + 255) & ~(size_t)255; return p; };

  __bf16* W1f  = (__bf16*)carve(160 * 64 * 2);
  __bf16* W2f  = (__bf16*)carve(64 * 64 * 2);
  __bf16* W3f  = (__bf16*)carve(64 * 64 * 2);
  __bf16* aW1f = (__bf16*)carve(160 * 64 * 2);
  __bf16* aW2f = (__bf16*)carve(64 * 16 * 2);
  __bf16* Wihf = (__bf16*)carve(64 * 192 * 2);
  __bf16* Whhf = (__bf16*)carve(64 * 192 * 2);
  __bf16* oW1f = (__bf16*)carve(96 * 64 * 2);
  __bf16* oW2f = (__bf16*)carve(64 * 64 * 2);
  __bf16* oW3f = (__bf16*)carve(64 * 16 * 2);
  float* stateA  = (float*)carve((size_t)NODES * 64 * 4);
  float* stateB  = (float*)carve((size_t)NODES * 64 * 4);
  float* msg_agg = (float*)carve((size_t)NODES * 64 * 4);
  float* att_sum = (float*)carve((size_t)NODES * 4);
  float* att_buf = (float*)carve((size_t)EDGES * 4);
  __bf16* msg_buf = (__bf16*)carve((size_t)EDGES * 64 * 2);

  auto pack = [&](const float* src, __bf16* dst, int Kr, int Nr, int KP, int NP) {
    hipLaunchKernelGGL(pack_b_frags, dim3((KP / 32) * (NP / 16)), dim3(512), 0, stream,
                       src, dst, Kr, Nr, KP, NP);
  };
  pack(msg_W1, W1f, 136, 64, 160, 64);
  pack(msg_W2, W2f, 64, 64, 64, 64);
  pack(msg_W3, W3f, 64, 64, 64, 64);
  pack(att_W1, aW1f, 136, 64, 160, 64);
  pack(att_W2, aW2f, 64, 1, 64, 16);
  pack(gru_Wih, Wihf, 64, 192, 64, 192);
  pack(gru_Whh, Whhf, 64, 192, 64, 192);
  pack(out_W1, oW1f, 66, 64, 96, 64);
  pack(out_W2, oW2f, 64, 64, 64, 64);
  pack(out_W3, oW3f, 64, 2, 64, 16);

  hipMemsetAsync(stateA, 0, (size_t)NODES * 64 * 4, stream);

  float* s = stateA;
  float* sn = stateB;
  const int edge_blocks = (EDGES + 127) / 128;
  const int node_blocks = (NODES + 127) / 128;
  const int scat_blocks = (int)(((size_t)EDGES * 64 + 255) / 256);

  for (int step = 0; step < 3; step++) {
    hipMemsetAsync(att_sum, 0, (size_t)NODES * 4, stream);
    hipMemsetAsync(msg_agg, 0, (size_t)NODES * 64 * 4, stream);

    hipLaunchKernelGGL(edge_kernel, dim3(edge_blocks), dim3(256), 0, stream,
                       s, b, J_msg, mnode, W1f, msg_b1, W2f, msg_b2, W3f, msg_b3,
                       aW1f, att_b1, aW2f, att_b2, msg_buf, att_buf, att_sum);

    hipLaunchKernelGGL(scatter_kernel, dim3(scat_blocks), dim3(256), 0, stream,
                       msg_buf, att_buf, att_sum, mnode, msg_agg);

    hipLaunchKernelGGL(node_kernel, dim3(node_blocks), dim3(256), 0, stream,
                       s, sn, msg_agg, b, Wihf, gru_bih, Whhf, gru_bhh,
                       oW1f, out_b1, oW2f, out_b2, oW3f, out_b3,
                       (float*)d_out, step);

    float* t = s; s = sn; sn = t;
  }
}
